// Net_19782619365956
// MI455X (gfx1250) — matrix operand order repaired and checked by reading, not compiled, not run on hardware
//
#include <hip/hip_runtime.h>
#include <stdint.h>

#define NATOM 262144
#define NCLQ  65536
#define NEDGE 524288
#define NPAIR 524288
#define NTREE 262144
#define NBAT  8192
#define HD    128
#define HD2   256
#define NLAY  3
#define NBOND 16

typedef __attribute__((ext_vector_type(16))) __bf16 bf16x16;
typedef __attribute__((ext_vector_type(8)))  float  f32x8;

static __device__ __forceinline__ float bf2f(unsigned short h) {
  union { unsigned u; float f; } c; c.u = ((unsigned)h) << 16; return c.f;
}
static __device__ __forceinline__ unsigned short f2bf(float f) {
  union { float f; unsigned u; } c; c.f = f;
  unsigned r = (c.u + 0x7FFFu + ((c.u >> 16) & 1u)) >> 16;
  return (unsigned short)r;
}
static __device__ __forceinline__ unsigned pack2bf(float a, float b) {
  return (unsigned)f2bf(a) | ((unsigned)f2bf(b) << 16);
}

// ---------------------------------------------------------------------------
// WMMA GEMM: C[M,Nw] = op(A)[M,K] x W[K,Nw](bf16)  (+ epilogue variants)
// AM: 0 = A f32 plain; 1 = A f32 * (1/max(rowcnt,1)); 2 = A bf16 with fused
//     per-column BN affine + ReLU (s,t arrays indexed by k).
// EM: 0 = store f32; 1 = store f32 + bias; 2 = C += relu(acc + bias);
//     3 = store bf16.
// Block: 256 thr (8 waves as 4x2); each wave computes 16x32 (2 wmma tiles
// sharing one A fragment) -> 64x64 block tile, 2 v_wmma per k-step per wave.
// ---------------------------------------------------------------------------
template <int AM, int EM>
__global__ __launch_bounds__(256) void gemm_wmma_k(
    const void* __restrict__ Aptr, const unsigned short* __restrict__ W,
    const float* __restrict__ bias, const float* __restrict__ rowcnt,
    const float* __restrict__ bnS, const float* __restrict__ bnT,
    float* __restrict__ Cf, unsigned short* __restrict__ Cb,
    int M, int K, int Nw) {
  __shared__ unsigned short As[64 * 32];   // A tile, bf16, [m][k]
  __shared__ unsigned short Ws[64 * 32];   // W tile, bf16, transposed [n][k]

  const int tid  = threadIdx.x;
  const int lane = tid & 31;
  const int wave = tid >> 5;
  const int wr   = wave >> 1;   // 0..3
  const int wc   = wave & 1;    // 0..1
  const int row0 = blockIdx.x * 64;
  const int col0 = blockIdx.y * 64;
  const int half = lane >> 4;
  const int mn   = lane & 15;

  f32x8 acc0, acc1;
#pragma unroll
  for (int i = 0; i < 8; ++i) { acc0[i] = 0.0f; acc1[i] = 0.0f; }

  for (int k0 = 0; k0 < K; k0 += 32) {
    // ---- stage A tile (64 rows x 32 k) as bf16, packed 64-bit LDS stores ----
    {
      const int r  = tid >> 3;          // 0..31
      const int kk = (tid & 7) * 4;     // 0..28
#pragma unroll
      for (int rr = r; rr < 64; rr += 32) {
        const int grow = row0 + rr;
        uint2 pk;
        if (AM == 2) {
          const unsigned short* Ab = (const unsigned short*)Aptr;
          uint2 raw = *(const uint2*)(Ab + (size_t)grow * K + k0 + kk);
          float v0 = bf2f((unsigned short)(raw.x & 0xFFFFu));
          float v1 = bf2f((unsigned short)(raw.x >> 16));
          float v2 = bf2f((unsigned short)(raw.y & 0xFFFFu));
          float v3 = bf2f((unsigned short)(raw.y >> 16));
          const int kc = k0 + kk;
          v0 = fmaxf(v0 * bnS[kc + 0] + bnT[kc + 0], 0.0f);
          v1 = fmaxf(v1 * bnS[kc + 1] + bnT[kc + 1], 0.0f);
          v2 = fmaxf(v2 * bnS[kc + 2] + bnT[kc + 2], 0.0f);
          v3 = fmaxf(v3 * bnS[kc + 3] + bnT[kc + 3], 0.0f);
          pk.x = pack2bf(v0, v1);
          pk.y = pack2bf(v2, v3);
        } else {
          const float* Af = (const float*)Aptr;
          float4 v4 = *(const float4*)(Af + (size_t)grow * K + k0 + kk);
          float sc = 1.0f;
          if (AM == 1) {
            float c = rowcnt[grow];
            sc = 1.0f / (c > 1.0f ? c : 1.0f);
          }
          pk.x = pack2bf(v4.x * sc, v4.y * sc);
          pk.y = pack2bf(v4.z * sc, v4.w * sc);
        }
        *(uint2*)&As[rr * 32 + kk] = pk;   // 8B aligned: kk multiple of 4
      }
      // ---- stage W tile (32 k x 64 n), store transposed [n][k] ----
      const int n4 = (tid & 15) * 4;      // 0..60
      const int kw = tid >> 4;            // 0..15
#pragma unroll
      for (int kk2 = kw; kk2 < 32; kk2 += 16) {
        uint2 w4 = *(const uint2*)(W + (size_t)(k0 + kk2) * Nw + col0 + n4);
        Ws[(n4 + 0) * 32 + kk2] = (unsigned short)(w4.x & 0xFFFFu);
        Ws[(n4 + 1) * 32 + kk2] = (unsigned short)(w4.x >> 16);
        Ws[(n4 + 2) * 32 + kk2] = (unsigned short)(w4.y & 0xFFFFu);
        Ws[(n4 + 3) * 32 + kk2] = (unsigned short)(w4.y >> 16);
      }
    }
    __syncthreads();

    // ---- build fragments per ISA 16-bit A/B wave32 layouts ----
    union Frag { bf16x16 v; unsigned u[8]; } afr, bf0, bf1;
    const int am  = wr * 16 + mn;         // A row within block tile
    const int bn0 = wc * 32 + mn;         // B column (first tile)
    const int bn1 = bn0 + 16;             // B column (second tile)
#pragma unroll
    for (int v = 0; v < 8; ++v) {
      const int ka = (v < 4 ? 2 * v : 8 + 2 * v) + 8 * half;  // even
      afr.u[v] = *(const unsigned*)(&As[am * 32 + ka]);
      const int kb = (v < 4 ? 2 * v : 8 + 2 * v) + 8 * half;  // even
      bf0.u[v] = *(const unsigned*)(&Ws[bn0 * 32 + kb]);
      bf1.u[v] = *(const unsigned*)(&Ws[bn1 * 32 + kb]);
    }
    acc0 = __builtin_amdgcn_wmma_f32_16x16x32_bf16(false, afr.v, false, bf0.v,
                                                   (short)0, acc0, false, false);
    acc1 = __builtin_amdgcn_wmma_f32_16x16x32_bf16(false, afr.v, false, bf1.v,
                                                   (short)0, acc1, false, false);
    __syncthreads();
  }

  // ---- epilogue: lane covers columns (mn, mn+16), rows v + 8*half ----
  const int ccol0 = col0 + wc * 32 + mn;
  const int ccol1 = ccol0 + 16;
#pragma unroll
  for (int v = 0; v < 8; ++v) {
    const int crow = row0 + wr * 16 + v + 8 * half;
    const size_t base = (size_t)crow * Nw;
    float v0 = acc0[v], v1 = acc1[v];
    if (EM == 3) {
      Cb[base + ccol0] = f2bf(v0);
      Cb[base + ccol1] = f2bf(v1);
    } else if (EM == 2) {
      v0 = fmaxf(v0 + bias[ccol0], 0.0f);
      v1 = fmaxf(v1 + bias[ccol1], 0.0f);
      Cf[base + ccol0] += v0;
      Cf[base + ccol1] += v1;
    } else if (EM == 1) {
      Cf[base + ccol0] = v0 + bias[ccol0];
      Cf[base + ccol1] = v1 + bias[ccol1];
    } else {
      Cf[base + ccol0] = v0;
      Cf[base + ccol1] = v1;
    }
  }
}

// ---------------------------------------------------------------------------
// GINE message: agg[dst] += relu(x[src] + bond_emb[attr])  (1 wave / edge)
// ---------------------------------------------------------------------------
__global__ void gine_msg_k(const float* __restrict__ X, const int* __restrict__ src,
                           const int* __restrict__ dst, const int* __restrict__ attr,
                           const float* __restrict__ emb, float* __restrict__ agg,
                           int E) {
  const int lane = threadIdx.x & 31, wave = threadIdx.x >> 5;
  for (int e = blockIdx.x * 8 + wave; e < E; e += gridDim.x * 8) {
    const int s = src[e], d = dst[e], a = attr[e];
    const float4 xv = *(const float4*)(X + (size_t)s * HD + lane * 4);
    const float4 ev = *(const float4*)(emb + (size_t)a * HD + lane * 4);
    float* o = agg + (size_t)d * HD + lane * 4;
    atomicAdd(o + 0, fmaxf(xv.x + ev.x, 0.0f));
    atomicAdd(o + 1, fmaxf(xv.y + ev.y, 0.0f));
    atomicAdd(o + 2, fmaxf(xv.z + ev.z, 0.0f));
    atomicAdd(o + 3, fmaxf(xv.w + ev.w, 0.0f));
  }
}

// out[dst] += X[src] (src==nullptr -> identity), optional per-dst count
__global__ void scatter_rows_k(const float* __restrict__ X, const int* __restrict__ src,
                               const int* __restrict__ dst, float* __restrict__ out,
                               float* __restrict__ cnt, int P) {
  const int lane = threadIdx.x & 31, wave = threadIdx.x >> 5;
  for (int p = blockIdx.x * 8 + wave; p < P; p += gridDim.x * 8) {
    const int s = src ? src[p] : p;
    const int d = dst[p];
    const float4 v = *(const float4*)(X + (size_t)s * HD + lane * 4);
    float* o = out + (size_t)d * HD + lane * 4;
    atomicAdd(o + 0, v.x);
    atomicAdd(o + 1, v.y);
    atomicAdd(o + 2, v.z);
    atomicAdd(o + 3, v.w);
    if (cnt && lane == 0) atomicAdd(&cnt[d], 1.0f);
  }
}

// Y += (1 + eps[0]) * X   (float4 vectorized)
__global__ void axpy_eps_k(float* __restrict__ Y, const float* __restrict__ X,
                           const float* __restrict__ eps, int n4) {
  const float a = 1.0f + eps[0];
  for (int i = blockIdx.x * blockDim.x + threadIdx.x; i < n4;
       i += gridDim.x * blockDim.x) {
    float4 y = ((float4*)Y)[i];
    const float4 x = ((const float4*)X)[i];
    y.x += a * x.x; y.y += a * x.y; y.z += a * x.z; y.w += a * x.w;
    ((float4*)Y)[i] = y;
  }
}

// per-column sum / sumsq over [M,C] (thread owns a column -> coalesced)
__global__ void colstats_f32_k(const float* __restrict__ X, float* __restrict__ sum,
                               float* __restrict__ sq, int M, int C) {
  const int rpt = blockDim.x / C;
  const int col = threadIdx.x % C;
  const int rs  = threadIdx.x / C;
  float s = 0.0f, q = 0.0f;
  for (int r = blockIdx.x * rpt + rs; r < M; r += gridDim.x * rpt) {
    const float v = X[(size_t)r * C + col];
    s += v; q += v * v;
  }
  atomicAdd(&sum[col], s);
  atomicAdd(&sq[col], q);
}
__global__ void colstats_bf16_k(const unsigned short* __restrict__ X,
                                float* __restrict__ sum, float* __restrict__ sq,
                                int M, int C) {
  const int rpt = blockDim.x / C;
  const int col = threadIdx.x % C;
  const int rs  = threadIdx.x / C;
  float s = 0.0f, q = 0.0f;
  for (int r = blockIdx.x * rpt + rs; r < M; r += gridDim.x * rpt) {
    const float v = bf2f(X[(size_t)r * C + col]);
    s += v; q += v * v;
  }
  atomicAdd(&sum[col], s);
  atomicAdd(&sq[col], q);
}

// S = g*rsqrt(var+1e-5), T = b - mu*S   (training-mode biased variance)
__global__ void bn_finalize_k(const float* __restrict__ sum, const float* __restrict__ sq,
                              const float* __restrict__ g, const float* __restrict__ b,
                              float* __restrict__ S, float* __restrict__ T, float invM) {
  const int c = threadIdx.x;
  const float mu = sum[c] * invM;
  const float var = sq[c] * invM - mu * mu;
  const float s = g[c] * rsqrtf(var + 1e-5f);
  S[c] = s;
  T[c] = b[c] - mu * s;
}

// X = relu(X * S[col] + T[col]) in place, C power of two
__global__ void normrelu_k(float* __restrict__ X, const float* __restrict__ S,
                           const float* __restrict__ T, int n4, int cm /* C/4-1 */) {
  for (int i = blockIdx.x * blockDim.x + threadIdx.x; i < n4;
       i += gridDim.x * blockDim.x) {
    const int j = (i & cm) << 2;
    float4 v = ((float4*)X)[i];
    v.x = fmaxf(v.x * S[j + 0] + T[j + 0], 0.0f);
    v.y = fmaxf(v.y * S[j + 1] + T[j + 1], 0.0f);
    v.z = fmaxf(v.z * S[j + 2] + T[j + 2], 0.0f);
    v.w = fmaxf(v.w * S[j + 3] + T[j + 3], 0.0f);
    ((float4*)X)[i] = v;
  }
}

__global__ void cvt_bf16_k(const float* __restrict__ in, unsigned short* __restrict__ out,
                           int n) {
  for (int i = blockIdx.x * blockDim.x + threadIdx.x; i < n;
       i += gridDim.x * blockDim.x)
    out[i] = f2bf(in[i]);
}

// out[b] = dot(relu(XG[b]+XCG[b]), linW) + linB[0]   (1 wave / row)
__global__ void final_head_k(const float* __restrict__ XG, const float* __restrict__ XCG,
                             const float* __restrict__ linW, const float* __restrict__ linB,
                             float* __restrict__ out, int Bn) {
  const int lane = threadIdx.x & 31, wave = threadIdx.x >> 5;
  for (int b = blockIdx.x * 8 + wave; b < Bn; b += gridDim.x * 8) {
    const float4 g = *(const float4*)(XG + (size_t)b * HD + lane * 4);
    const float4 c = *(const float4*)(XCG + (size_t)b * HD + lane * 4);
    const float4 w = *(const float4*)(linW + lane * 4);
    float a = fmaxf(g.x + c.x, 0.0f) * w.x + fmaxf(g.y + c.y, 0.0f) * w.y +
              fmaxf(g.z + c.z, 0.0f) * w.z + fmaxf(g.w + c.w, 0.0f) * w.w;
    for (int off = 16; off; off >>= 1) a += __shfl_down(a, off, 32);
    if (lane == 0) out[b] = a + linB[0];
  }
}

// ---------------------------------------------------------------------------
extern "C" void kernel_launch(void* const* d_in, const int* in_sizes, int n_in,
                              void* d_out, int out_size, void* d_ws, size_t ws_size,
                              hipStream_t stream) {
  (void)in_sizes; (void)n_in; (void)out_size; (void)ws_size;

  // ---- inputs ----
  const float* x          = (const float*)d_in[0];
  const float* x_clique   = (const float*)d_in[1];
  const int*   edge_attr  = (const int*)d_in[2];
  const int*   edge_src   = (const int*)d_in[3];
  const int*   edge_dst   = (const int*)d_in[4];
  const int*   a2c_row    = (const int*)d_in[5];
  const int*   a2c_col    = (const int*)d_in[6];
  const int*   tree_src   = (const int*)d_in[7];
  const int*   tree_dst   = (const int*)d_in[8];
  const int*   batch      = (const int*)d_in[9];
  const int*   tree_batch = (const int*)d_in[10];
  const float* bond_emb   = (const float*)d_in[11];
  const float* gine_eps   = (const float*)d_in[12];
  const float* gine_W1    = (const float*)d_in[13];
  const float* gine_bn_g  = (const float*)d_in[15];
  const float* gine_bn_b  = (const float*)d_in[16];
  const float* gine_W2    = (const float*)d_in[17];
  const float* atom_bn_g  = (const float*)d_in[19];
  const float* atom_bn_b  = (const float*)d_in[20];
  const float* a2c_W      = (const float*)d_in[21];
  const float* a2c_b      = (const float*)d_in[22];
  const float* gin_eps    = (const float*)d_in[23];
  const float* gin_W1     = (const float*)d_in[24];
  const float* gin_bn_g   = (const float*)d_in[26];
  const float* gin_bn_b   = (const float*)d_in[27];
  const float* gin_W2     = (const float*)d_in[28];
  const float* clique_bn_g= (const float*)d_in[30];
  const float* clique_bn_b= (const float*)d_in[31];
  const float* c2a_W      = (const float*)d_in[32];
  const float* c2a_b      = (const float*)d_in[33];
  const float* atom_lin_W = (const float*)d_in[34];
  const float* atom_lin_b = (const float*)d_in[35];
  const float* clique_lin_W=(const float*)d_in[36];
  const float* clique_lin_b=(const float*)d_in[37];
  const float* lin_W      = (const float*)d_in[38];
  const float* lin_b      = (const float*)d_in[39];
  float* out = (float*)d_out;

  // ---- workspace layout ----
  char* ws = (char*)d_ws;
  size_t o = 0;
  auto take = [&](size_t bytes) { char* p = ws + o; o += (bytes + 255) & ~(size_t)255; return p; };
  float* XA    = (float*)take((size_t)NATOM * HD * 4);
  float* XC    = (float*)take((size_t)NCLQ * HD * 4);
  float* AGGA  = (float*)take((size_t)NATOM * HD * 4);
  float* AGGC  = (float*)take((size_t)NCLQ * HD * 4);
  unsigned short* T1 = (unsigned short*)take((size_t)NATOM * HD2 * 2);
  float* CNT_N = (float*)take((size_t)NATOM * 4);
  float* CNT_C = (float*)take((size_t)NCLQ * 4);
  float* POOLA = (float*)take((size_t)NBAT * HD * 4);
  float* POOLC = (float*)take((size_t)NBAT * HD * 4);
  float* XG    = (float*)take((size_t)NBAT * HD * 4);
  float* XCG   = (float*)take((size_t)NBAT * HD * 4);
  float* STATS = (float*)take(4 * HD2 * 4);  // sum | sumsq | S | T
  float* Ssum = STATS, *Ssq = STATS + HD2, *Sv = STATS + 2 * HD2, *Tv = STATS + 3 * HD2;
  unsigned short* WB_gineW1 = (unsigned short*)take((size_t)NLAY * HD * HD2 * 2);
  unsigned short* WB_gineW2 = (unsigned short*)take((size_t)NLAY * HD2 * HD * 2);
  unsigned short* WB_ginW1  = (unsigned short*)take((size_t)NLAY * HD * HD2 * 2);
  unsigned short* WB_ginW2  = (unsigned short*)take((size_t)NLAY * HD2 * HD * 2);
  unsigned short* WB_a2c    = (unsigned short*)take((size_t)NLAY * HD * HD * 2);
  unsigned short* WB_c2a    = (unsigned short*)take((size_t)NLAY * HD * HD * 2);
  unsigned short* WB_alin   = (unsigned short*)take((size_t)HD * HD * 2);
  unsigned short* WB_clin   = (unsigned short*)take((size_t)HD * HD * 2);

  // ---- init: feature copies + one-time bf16 weight conversion ----
  hipMemcpyAsync(XA, x, (size_t)NATOM * HD * 4, hipMemcpyDeviceToDevice, stream);
  hipMemcpyAsync(XC, x_clique, (size_t)NCLQ * HD * 4, hipMemcpyDeviceToDevice, stream);
  cvt_bf16_k<<<256, 256, 0, stream>>>(gine_W1, WB_gineW1, NLAY * HD * HD2);
  cvt_bf16_k<<<256, 256, 0, stream>>>(gine_W2, WB_gineW2, NLAY * HD2 * HD);
  cvt_bf16_k<<<256, 256, 0, stream>>>(gin_W1, WB_ginW1, NLAY * HD * HD2);
  cvt_bf16_k<<<256, 256, 0, stream>>>(gin_W2, WB_ginW2, NLAY * HD2 * HD);
  cvt_bf16_k<<<256, 256, 0, stream>>>(a2c_W, WB_a2c, NLAY * HD * HD);
  cvt_bf16_k<<<256, 256, 0, stream>>>(c2a_W, WB_c2a, NLAY * HD * HD);
  cvt_bf16_k<<<64, 256, 0, stream>>>(atom_lin_W, WB_alin, HD * HD);
  cvt_bf16_k<<<64, 256, 0, stream>>>(clique_lin_W, WB_clin, HD * HD);

  const dim3 blk(256);
  for (int i = 0; i < NLAY; ++i) {
    // === GINE on atoms ===
    hipMemsetAsync(AGGA, 0, (size_t)NATOM * HD * 4, stream);
    gine_msg_k<<<NEDGE / 8, blk, 0, stream>>>(XA, edge_src, edge_dst, edge_attr,
                                              bond_emb + (size_t)i * NBOND * HD, AGGA, NEDGE);
    axpy_eps_k<<<2048, blk, 0, stream>>>(AGGA, XA, gine_eps + i, NATOM * HD / 4);
    // MLP1: T1 = AGGA @ W1 (bias dropped: cancels in BN)
    gemm_wmma_k<0, 3><<<dim3(NATOM / 64, HD2 / 64), blk, 0, stream>>>(
        AGGA, WB_gineW1 + (size_t)i * HD * HD2, nullptr, nullptr, nullptr, nullptr,
        nullptr, T1, NATOM, HD, HD2);
    hipMemsetAsync(STATS, 0, 2 * HD2 * 4, stream);
    colstats_bf16_k<<<256, blk, 0, stream>>>(T1, Ssum, Ssq, NATOM, HD2);
    bn_finalize_k<<<1, HD2, 0, stream>>>(Ssum, Ssq, gine_bn_g + (size_t)i * HD2,
                                         gine_bn_b + (size_t)i * HD2, Sv, Tv, 1.0f / NATOM);
    // MLP2: XA = relu(BN(T1)) @ W2  (BN affine+relu fused in A-load; b2 cancels in atom BN)
    gemm_wmma_k<2, 0><<<dim3(NATOM / 64, HD / 64), blk, 0, stream>>>(
        T1, WB_gineW2 + (size_t)i * HD2 * HD, nullptr, nullptr, Sv, Tv,
        XA, nullptr, NATOM, HD2, HD);
    // atom BN + relu
    hipMemsetAsync(STATS, 0, 2 * HD2 * 4, stream);
    colstats_f32_k<<<256, blk, 0, stream>>>(XA, Ssum, Ssq, NATOM, HD);
    bn_finalize_k<<<1, HD, 0, stream>>>(Ssum, Ssq, atom_bn_g + (size_t)i * HD,
                                        atom_bn_b + (size_t)i * HD, Sv, Tv, 1.0f / NATOM);
    normrelu_k<<<2048, blk, 0, stream>>>(XA, Sv, Tv, NATOM * HD / 4, HD / 4 - 1);
    // === atom -> clique: mean + linear, accumulate into XC ===
    hipMemsetAsync(AGGC, 0, (size_t)NCLQ * HD * 4, stream);
    hipMemsetAsync(CNT_C, 0, (size_t)NCLQ * 4, stream);
    scatter_rows_k<<<NPAIR / 8, blk, 0, stream>>>(XA, a2c_row, a2c_col, AGGC, CNT_C, NPAIR);
    gemm_wmma_k<1, 2><<<dim3(NCLQ / 64, HD / 64), blk, 0, stream>>>(
        AGGC, WB_a2c + (size_t)i * HD * HD, a2c_b + (size_t)i * HD, CNT_C,
        nullptr, nullptr, XC, nullptr, NCLQ, HD, HD);
    // === GIN on clique tree ===
    hipMemsetAsync(AGGC, 0, (size_t)NCLQ * HD * 4, stream);
    scatter_rows_k<<<NTREE / 8, blk, 0, stream>>>(XC, tree_src, tree_dst, AGGC, nullptr, NTREE);
    axpy_eps_k<<<2048, blk, 0, stream>>>(AGGC, XC, gin_eps + i, NCLQ * HD / 4);
    gemm_wmma_k<0, 3><<<dim3(NCLQ / 64, HD2 / 64), blk, 0, stream>>>(
        AGGC, WB_ginW1 + (size_t)i * HD * HD2, nullptr, nullptr, nullptr, nullptr,
        nullptr, T1, NCLQ, HD, HD2);
    hipMemsetAsync(STATS, 0, 2 * HD2 * 4, stream);
    colstats_bf16_k<<<256, blk, 0, stream>>>(T1, Ssum, Ssq, NCLQ, HD2);
    bn_finalize_k<<<1, HD2, 0, stream>>>(Ssum, Ssq, gin_bn_g + (size_t)i * HD2,
                                         gin_bn_b + (size_t)i * HD2, Sv, Tv, 1.0f / NCLQ);
    gemm_wmma_k<2, 0><<<dim3(NCLQ / 64, HD / 64), blk, 0, stream>>>(
        T1, WB_ginW2 + (size_t)i * HD2 * HD, nullptr, nullptr, Sv, Tv,
        XC, nullptr, NCLQ, HD2, HD);
    hipMemsetAsync(STATS, 0, 2 * HD2 * 4, stream);
    colstats_f32_k<<<256, blk, 0, stream>>>(XC, Ssum, Ssq, NCLQ, HD);
    bn_finalize_k<<<1, HD, 0, stream>>>(Ssum, Ssq, clique_bn_g + (size_t)i * HD,
                                        clique_bn_b + (size_t)i * HD, Sv, Tv, 1.0f / NCLQ);
    normrelu_k<<<2048, blk, 0, stream>>>(XC, Sv, Tv, NCLQ * HD / 4, HD / 4 - 1);
    // === clique -> atom: mean + linear, accumulate into XA ===
    hipMemsetAsync(AGGA, 0, (size_t)NATOM * HD * 4, stream);
    hipMemsetAsync(CNT_N, 0, (size_t)NATOM * 4, stream);
    scatter_rows_k<<<NPAIR / 8, blk, 0, stream>>>(XC, a2c_col, a2c_row, AGGA, CNT_N, NPAIR);
    gemm_wmma_k<1, 2><<<dim3(NATOM / 64, HD / 64), blk, 0, stream>>>(
        AGGA, WB_c2a + (size_t)i * HD * HD, c2a_b + (size_t)i * HD, CNT_N,
        nullptr, nullptr, XA, nullptr, NATOM, HD, HD);
  }

  // === readout: mean pool per graph, two linears, relu-sum, final dot ===
  hipMemsetAsync(POOLA, 0, (size_t)NBAT * HD * 4, stream);
  hipMemsetAsync(POOLC, 0, (size_t)NBAT * HD * 4, stream);
  hipMemsetAsync(CNT_N, 0, (size_t)NBAT * 4, stream);
  hipMemsetAsync(CNT_C, 0, (size_t)NBAT * 4, stream);
  scatter_rows_k<<<NATOM / 8, blk, 0, stream>>>(XA, nullptr, batch, POOLA, CNT_N, NATOM);
  scatter_rows_k<<<NCLQ / 8, blk, 0, stream>>>(XC, nullptr, tree_batch, POOLC, CNT_C, NCLQ);
  gemm_wmma_k<1, 1><<<dim3(NBAT / 64, HD / 64), blk, 0, stream>>>(
      POOLA, WB_alin, atom_lin_b, CNT_N, nullptr, nullptr, XG, nullptr, NBAT, HD, HD);
  gemm_wmma_k<1, 1><<<dim3(NBAT / 64, HD / 64), blk, 0, stream>>>(
      POOLC, WB_clin, clique_lin_b, CNT_C, nullptr, nullptr, XCG, nullptr, NBAT, HD, HD);
  final_head_k<<<NBAT / 8, blk, 0, stream>>>(XG, XCG, lin_W, lin_b, out, NBAT);
}
